// ContrastiveQueue_55714315764155
// MI455X (gfx1250) — matrix operand order repaired and checked by reading, not compiled, not run on hardware
//
#include <hip/hip_runtime.h>

typedef __attribute__((ext_vector_type(16))) _Float16 v16h;
typedef __attribute__((ext_vector_type(8)))  _Float16 v8h;
typedef __attribute__((ext_vector_type(8)))  float    v8f;

#define N_ROWS   4096
#define OB_DIM   64
#define HID      256
#define C_DIM    128
#define K_DIM    32768
#define KTILES   (K_DIM / 16)         // 2048 column tiles
#define KT_HALF  (KTILES / 2)         // 1024 per K-half
// exp(x/T) = exp2(x * (1/T) * log2(e)); 1/T = 5. Folded into q at f16 convert.
#define EXP2_SCALE 7.21347520444481703680f

// ---------------------------------------------------------------------------
// Kernel 1: MLP (tanh, tanh, linear) + L2-normalize rows; store
// q * EXP2_SCALE as f16 so the WMMA accumulator is directly the exp2 arg.
// ---------------------------------------------------------------------------
__global__ __launch_bounds__(256) void ContrastiveQueue_mlp_kernel(
    const float* __restrict__ ob,    // [4096,64]
    const float* __restrict__ W0,    // [64,256]
    const float* __restrict__ b0,    // [256]
    const float* __restrict__ W1,    // [256,256]
    const float* __restrict__ b1,    // [256]
    const float* __restrict__ Wout,  // [256,128]
    const float* __restrict__ bout,  // [128]
    _Float16* __restrict__ qf16)     // [4096,128] row-major, pre-scaled
{
    __shared__ float s_ob[OB_DIM];
    __shared__ float s_h0[HID];
    __shared__ float s_h1[HID];
    __shared__ float s_red[C_DIM];

    const int n = blockIdx.x;
    const int t = threadIdx.x;

    if (t < OB_DIM) s_ob[t] = ob[(size_t)n * OB_DIM + t];
    __syncthreads();

    {
        float acc = b0[t];
        #pragma unroll 8
        for (int i = 0; i < OB_DIM; ++i) acc = fmaf(s_ob[i], W0[i * HID + t], acc);
        s_h0[t] = tanhf(acc);
    }
    __syncthreads();

    {
        float acc = b1[t];
        #pragma unroll 8
        for (int i = 0; i < HID; ++i) acc = fmaf(s_h0[i], W1[i * HID + t], acc);
        s_h1[t] = tanhf(acc);
    }
    __syncthreads();

    float qv = 0.0f;
    if (t < C_DIM) {
        float acc = bout[t];
        #pragma unroll 8
        for (int i = 0; i < HID; ++i) acc = fmaf(s_h1[i], Wout[i * C_DIM + t], acc);
        qv = acc;
        s_red[t] = acc * acc;
    }
    __syncthreads();

    for (int off = C_DIM / 2; off > 0; off >>= 1) {
        if (t < off) s_red[t] += s_red[t + off];
        __syncthreads();
    }

    if (t < C_DIM) {
        float inv = EXP2_SCALE / fmaxf(sqrtf(s_red[0]), 1e-12f);
        qf16[(size_t)n * C_DIM + t] = (_Float16)(qv * inv);
    }
}

// ---------------------------------------------------------------------------
// Kernel 2: repack queue [128, 32768] f32 into WMMA-B f16 per-lane layout.
//   lanes 0-15  hold contraction rows c = ck*32 + 0..7,16..23 (2 per VGPR)
//   lanes 16-31 hold contraction rows c = ck*32 + 8..15,24..31
//   lane's N column = tile*16 + (lane & 15)
// ---------------------------------------------------------------------------
__global__ __launch_bounds__(128) void ContrastiveQueue_packB_kernel(
    const float* __restrict__ queue,  // [128, 32768]
    _Float16* __restrict__ packB)
{
    const int tile = blockIdx.x;
    const int lane = threadIdx.x & 31;
    const int ck   = threadIdx.x >> 5;          // 0..3
    const int kcol  = tile * 16 + (lane & 15);
    const int cbase = ck * 32 + ((lane >> 4) ? 8 : 0);

    _Float16* dst = packB + (((size_t)(tile * 4 + ck) * 32 + lane) * 16);
    #pragma unroll
    for (int j = 0; j < 16; ++j)
        dst[j] = (_Float16)queue[(size_t)(cbase + j + (j >= 8 ? 8 : 0)) * K_DIM + kcol];
}

// ---------------------------------------------------------------------------
// Kernel 3: partial sum-of-exp of logits.
// Grid: (128 row-groups of 32 rows) x (2 K-halves). One block = 8 waves.
// Each wave holds TWO A row-tile sets in registers; every streamed B tile
// feeds 2 independent WMMA chains (2x B reuse, 2x WMMA issue ILP).
// Accumulator is pre-scaled -> epilogue is bare v_exp_f32 + v_pk_add_f32.
// ---------------------------------------------------------------------------
__global__ __launch_bounds__(256) void ContrastiveQueue_lse_kernel(
    const _Float16* __restrict__ qf16,   // [4096,128] pre-scaled
    const _Float16* __restrict__ packB,  // packed B tiles
    float* __restrict__ pws)             // [2][4096] partial sum-of-exp
{
    const int rowgrp = blockIdx.x;       // 0..127  (32 rows each)
    const int khalf  = blockIdx.y;       // 0..1
    const int tid  = threadIdx.x;
    const int wave = tid >> 5;           // 0..7
    const int lane = tid & 31;
    const int lrow = lane & 15;
    const int grp  = lane >> 4;          // 0 or 1

    // ---- A tiles (ISA 16-bit A layout: lane<16 -> K 0..7,16..23;
    //      lane>=16 -> K 8..15,24..31); 4 tiles cover c = 0..127
    v16h a0[4], a1[4];
    {
        const _Float16* qr0 = qf16 + (size_t)(rowgrp * 32 + lrow) * C_DIM;
        const _Float16* qr1 = qr0 + 16 * C_DIM;
        #pragma unroll
        for (int kk = 0; kk < 4; ++kk) {
            const int c0 = kk * 32 + grp * 8;
            const int c1 = kk * 32 + 16 + grp * 8;
            union { v16h v; v8h h[2]; } av;
            av.h[0] = *(const v8h*)(qr0 + c0);
            av.h[1] = *(const v8h*)(qr0 + c1);
            a0[kk] = av.v;
            av.h[0] = *(const v8h*)(qr1 + c0);
            av.h[1] = *(const v8h*)(qr1 + c1);
            a1[kk] = av.v;
        }
    }

    float se0[8] = {0.f,0.f,0.f,0.f,0.f,0.f,0.f,0.f};
    float se1[8] = {0.f,0.f,0.f,0.f,0.f,0.f,0.f,0.f};

    const v16h* pb = (const v16h*)packB;
    const int tEnd = (khalf + 1) * KT_HALF;
    for (int t = khalf * KT_HALF + wave; t < tEnd; t += 8) {
        const v16h* bt = pb + ((size_t)t * 128 + lane);   // 128 v16h per tile
        __builtin_prefetch((const void*)(bt + 8 * 128), 0, 0);  // next tile
        v8f acc0 = {}, acc1 = {};
        #pragma unroll
        for (int ck = 0; ck < 4; ++ck) {
            v16h b = bt[ck * 32];
            acc0 = __builtin_amdgcn_wmma_f32_16x16x32_f16(
                false, a0[ck], false, b, (short)0, acc0, false, false);
            acc1 = __builtin_amdgcn_wmma_f32_16x16x32_f16(
                false, a1[ck], false, b, (short)0, acc1, false, false);
        }
        // D layout: acc[j] -> row M = j + grp*8 (of its tile), col N = lane&15
        // |acc| <= EXP2_SCALE -> raw v_exp_f32 is safe (no range guard needed)
        #pragma unroll
        for (int j = 0; j < 8; ++j) {
            se0[j] += __builtin_amdgcn_exp2f(acc0[j]);
            se1[j] += __builtin_amdgcn_exp2f(acc1[j]);
        }
    }

    // butterfly-reduce across the 16 lanes that share each row
    #pragma unroll
    for (int j = 0; j < 8; ++j) {
        float v = se0[j];
        v += __shfl_xor(v, 1, 32);
        v += __shfl_xor(v, 2, 32);
        v += __shfl_xor(v, 4, 32);
        v += __shfl_xor(v, 8, 32);
        se0[j] = v;
        float w = se1[j];
        w += __shfl_xor(w, 1, 32);
        w += __shfl_xor(w, 2, 32);
        w += __shfl_xor(w, 4, 32);
        w += __shfl_xor(w, 8, 32);
        se1[j] = w;
    }

    __shared__ float partial[8][32];
    if ((lane & 15) == 0) {
        #pragma unroll
        for (int j = 0; j < 8; ++j) {
            partial[wave][grp * 8 + j]      = se0[j];   // rows 0..15 of group
            partial[wave][16 + grp * 8 + j] = se1[j];   // rows 16..31 of group
        }
    }
    __syncthreads();

    if (tid < 32) {
        float s = 0.f;
        #pragma unroll
        for (int w = 0; w < 8; ++w) s += partial[w][tid];
        pws[(size_t)khalf * N_ROWS + rowgrp * 32 + tid] = s;
    }
}

// ---------------------------------------------------------------------------
// Kernel 4: finalize — out[n] = log(p0[n] + p1[n])
// ---------------------------------------------------------------------------
__global__ __launch_bounds__(256) void ContrastiveQueue_final_kernel(
    const float* __restrict__ pws,   // [2][4096]
    float* __restrict__ out)         // [4096]
{
    const int n = blockIdx.x * 256 + threadIdx.x;
    if (n < N_ROWS)
        out[n] = logf(pws[n] + pws[N_ROWS + n]);
}

// ---------------------------------------------------------------------------
extern "C" void kernel_launch(void* const* d_in, const int* in_sizes, int n_in,
                              void* d_out, int out_size, void* d_ws, size_t ws_size,
                              hipStream_t stream) {
    const float* ob    = (const float*)d_in[0];
    const float* W0    = (const float*)d_in[1];
    const float* b0    = (const float*)d_in[2];
    const float* W1    = (const float*)d_in[3];
    const float* b1    = (const float*)d_in[4];
    const float* Wout  = (const float*)d_in[5];
    const float* bout  = (const float*)d_in[6];
    const float* queue = (const float*)d_in[7];
    float* out = (float*)d_out;

    // workspace layout: qf16 (1 MB) | packB (8 MB) | partials (32 KB)
    char* ws = (char*)d_ws;
    _Float16* qf16  = (_Float16*)ws;
    _Float16* packB = (_Float16*)(ws + (size_t)N_ROWS * C_DIM * sizeof(_Float16));
    float*    pws   = (float*)(ws + (size_t)N_ROWS * C_DIM * sizeof(_Float16)
                                  + (size_t)K_DIM * C_DIM * sizeof(_Float16));

    ContrastiveQueue_mlp_kernel<<<N_ROWS, 256, 0, stream>>>(
        ob, W0, b0, W1, b1, Wout, bout, qf16);

    ContrastiveQueue_packB_kernel<<<KTILES, 128, 0, stream>>>(queue, packB);

    ContrastiveQueue_lse_kernel<<<dim3(N_ROWS / 32, 2), 256, 0, stream>>>(
        qf16, packB, pws);

    ContrastiveQueue_final_kernel<<<N_ROWS / 256, 256, 0, stream>>>(pws, out);
}
